// MoE_8589934592028
// MI455X (gfx1250) — compile-verified, matrix operand order checked by reading, not run on hardware
//
#include <hip/hip_runtime.h>
#include <math.h>

typedef __attribute__((ext_vector_type(16))) __bf16 v16bf;
typedef __attribute__((ext_vector_type(8)))  __bf16 v8bf;
typedef __attribute__((ext_vector_type(8)))  float  v8f;

#define N_TOK 8192
#define D_IN  1024
#define H_DIM 2048
#define O_DIM 1024
#define E_NUM 8

#define BN        32           // tokens per block
#define XS_STRIDE (D_IN + 8)   // bf16 elems; 2064B rows -> 16B aligned, bank-skewed
#define HC        128          // H chunk per double-stage iteration
#define HSS       (HC + 8)     // 272B rows

#define LDS_XS_BYTES (BN * XS_STRIDE * 2)          // 66048
#define LDS_HS_BYTES (BN * HSS * 2)                // 8704
#define LDS_GS_BYTES (BN * E_NUM * 4)              // 1024
#define LDS_TOTAL    (LDS_XS_BYTES + LDS_HS_BYTES + LDS_GS_BYTES)

// ---------------------------------------------------------------------------
// x: f32 -> bf16 (row-major, unchanged layout)
__global__ void convert_x_bf16(const float* __restrict__ x, __bf16* __restrict__ xb, long n) {
  long i = (long)blockIdx.x * blockDim.x + threadIdx.x;
  if (i < n) xb[i] = (__bf16)x[i];
}

// ---------------------------------------------------------------------------
// Pack weights [E, K, C] f32 into WMMA B-fragment order, bf16:
//   p = (((e*CT + ct)*KB + kb)*32 + lane)*16 + i
//   element(lane, i) = w[e][kb*32 + klocal(lane,i)][ct*16 + lane%16]
//   klocal = (i>=8 ? 16 : 0) + (lane/16)*8 + (i%8)
__global__ void pack_weights_bf16(const float* __restrict__ w, __bf16* __restrict__ out,
                                  int K, int C) {
  long p = (long)blockIdx.x * blockDim.x + threadIdx.x;
  long total = (long)E_NUM * K * C;
  if (p >= total) return;
  int  i    = (int)(p & 15);
  int  lane = (int)((p >> 4) & 31);
  long blk  = p >> 9;
  int  KB   = K / 32;
  int  CT   = C / 16;
  int  kb   = (int)(blk % KB);
  long r1   = blk / KB;
  int  ct   = (int)(r1 % CT);
  int  e    = (int)(r1 / CT);
  int  kl   = ((i >= 8) ? 16 : 0) + (lane >> 4) * 8 + (i & 7);
  int  row  = kb * 32 + kl;
  int  col  = ct * 16 + (lane & 15);
  out[p] = (__bf16)w[((long)e * K + row) * C + col];
}

// ---------------------------------------------------------------------------
// Gating: one wave32 per token; softmax over E=8 experts.
__global__ void gating_kernel(const float* __restrict__ x, const float* __restrict__ wg,
                              const float* __restrict__ bg, float* __restrict__ gate) {
  int wave = threadIdx.x >> 5;
  int lane = threadIdx.x & 31;
  int n = blockIdx.x * 8 + wave;

  float acc[E_NUM];
#pragma unroll
  for (int e = 0; e < E_NUM; ++e) acc[e] = 0.f;

  for (int j = lane; j < D_IN; j += 32) {
    float xv = x[(long)n * D_IN + j];
    const float4* wr = (const float4*)(wg + (long)j * E_NUM);
    float4 a = wr[0], b = wr[1];
    acc[0] += xv * a.x; acc[1] += xv * a.y; acc[2] += xv * a.z; acc[3] += xv * a.w;
    acc[4] += xv * b.x; acc[5] += xv * b.y; acc[6] += xv * b.z; acc[7] += xv * b.w;
  }
#pragma unroll
  for (int e = 0; e < E_NUM; ++e)
#pragma unroll
    for (int off = 16; off > 0; off >>= 1)
      acc[e] += __shfl_xor(acc[e], off, 32);

  if (lane == 0) {
    float l[E_NUM], mx = -1e30f;
#pragma unroll
    for (int e = 0; e < E_NUM; ++e) { l[e] = acc[e] + bg[e]; mx = fmaxf(mx, l[e]); }
    float s = 0.f;
#pragma unroll
    for (int e = 0; e < E_NUM; ++e) { l[e] = __expf(l[e] - mx); s += l[e]; }
    float inv = 1.f / s;
#pragma unroll
    for (int e = 0; e < E_NUM; ++e) gate[(long)n * E_NUM + e] = l[e] * inv;
  }
}

// ---------------------------------------------------------------------------
// Fused 2-layer expert MLP + gated combine.
// One block = 32 tokens, 16 waves arranged 2 (M) x 8 (N).
__launch_bounds__(512, 1)
__global__ void moe_ffn_kernel(const __bf16* __restrict__ xb,
                               const __bf16* __restrict__ w1p,
                               const __bf16* __restrict__ w2p,
                               const float* __restrict__ b1,
                               const float* __restrict__ b2,
                               const float* __restrict__ gate,
                               float* __restrict__ out) {
  extern __shared__ char smem[];
  __bf16* xs = (__bf16*)smem;                    // [BN][XS_STRIDE]
  __bf16* hs = (__bf16*)(smem + LDS_XS_BYTES);   // [BN][HSS]
  float*  gs = (float*)(smem + LDS_XS_BYTES + LDS_HS_BYTES);  // [BN][E_NUM]

  const int tid   = threadIdx.x;
  const int wave  = tid >> 5;
  const int lane  = tid & 31;
  const int wm    = wave >> 3;    // 0..1 : row-tile group of 16 tokens
  const int wn    = wave & 7;     // 0..7 : column strip of 128
  const int hlf   = lane >> 4;
  const int nloc  = lane & 15;
  const int tile0 = blockIdx.x * BN;

  // ---- stage x tile (BN x 1024 bf16) into LDS with async global->LDS copies
  for (int u = tid; u < BN * (D_IN / 8); u += 512) {
    int row = u >> 7;                 // / (D_IN/8)
    int c8  = u & 127;
    unsigned dst = (unsigned)(row * (XS_STRIDE * 2) + c8 * 16);   // LDS byte offset
    unsigned long long src = (unsigned long long)(xb + (long)(tile0 + row) * D_IN + c8 * 8);
    asm volatile("global_load_async_to_lds_b128 %0, %1, off"
                 :: "v"(dst), "v"(src) : "memory");
  }
  for (int u = tid; u < BN * E_NUM; u += 512)
    gs[u] = gate[(long)(tile0 + (u >> 3)) * E_NUM + (u & 7)];
  asm volatile("s_wait_asynccnt 0x0" ::: "memory");
  __syncthreads();

  v8f outacc[8];
#pragma unroll
  for (int t = 0; t < 8; ++t) outacc[t] = {};

  const int KB1 = D_IN / 32;    // 32
  const int CT1 = H_DIM / 16;   // 128
  const int KB2 = H_DIM / 32;   // 64
  const int CT2 = O_DIM / 16;   // 64

  for (int e = 0; e < E_NUM; ++e) {
    v8f eoacc[8];
#pragma unroll
    for (int t = 0; t < 8; ++t) eoacc[t] = {};

    for (int chunk = 0; chunk < H_DIM / HC; ++chunk) {
      // ---- stage 1: h tile = relu(x @ w1 + b1)
      // wave (wm, wn) computes rows wm*16..+15, cols chunk*128 + wn*16..+15
      const int ct1 = chunk * 8 + wn;
      const __bf16* bptr = w1p + ((((long)e * CT1 + ct1) * KB1) * 32 + lane) * 16;
      __builtin_prefetch(bptr, 0, 1);
      const int arow = wm * 16 + nloc;
      v8f acc1 = {};
#pragma unroll 4
      for (int kb = 0; kb < KB1; ++kb) {
        const int  k0  = kb * 32 + hlf * 8;
        const v8bf alo = *(const v8bf*)&xs[arow * XS_STRIDE + k0];
        const v8bf ahi = *(const v8bf*)&xs[arow * XS_STRIDE + k0 + 16];
        v16bf a;
#pragma unroll
        for (int q = 0; q < 8; ++q) { a[q] = alo[q]; a[q + 8] = ahi[q]; }
        v16bf b = *(const v16bf*)(bptr + (long)kb * 512);
        acc1 = __builtin_amdgcn_wmma_f32_16x16x32_bf16(false, a, false, b,
                                                       (short)0, acc1, false, false);
      }
      const float bias1 = b1[e * H_DIM + ct1 * 16 + nloc];
#pragma unroll
      for (int r = 0; r < 8; ++r) {
        int   m = wm * 16 + r + 8 * hlf;
        float v = acc1[r] + bias1;
        hs[m * HSS + wn * 16 + nloc] = (__bf16)(v > 0.f ? v : 0.f);
      }
      __syncthreads();

      // ---- stage 2: eoacc += h_chunk @ w2_chunk
      // wave (wm, wn) owns rows wm*16..+15, cols wn*128..+127
#pragma unroll
      for (int t = 0; t < 8; ++t) {
        const int ct2 = wn * 8 + t;
        const __bf16* b2ptr =
            w2p + (((long)e * CT2 + ct2) * KB2 + chunk * (HC / 32)) * 512 + lane * 16;
#pragma unroll
        for (int kbl = 0; kbl < HC / 32; ++kbl) {
          const int  k0  = kbl * 32 + hlf * 8;
          const v8bf alo = *(const v8bf*)&hs[arow * HSS + k0];
          const v8bf ahi = *(const v8bf*)&hs[arow * HSS + k0 + 16];
          v16bf a;
#pragma unroll
          for (int q = 0; q < 8; ++q) { a[q] = alo[q]; a[q + 8] = ahi[q]; }
          v16bf b = *(const v16bf*)(b2ptr + kbl * 512);
          eoacc[t] = __builtin_amdgcn_wmma_f32_16x16x32_bf16(false, a, false, b,
                                                             (short)0, eoacc[t], false, false);
        }
      }
      __syncthreads();
    }

    // ---- gated combine: out += g[n,e] * (eo + b2[e])
    float gw[8];
#pragma unroll
    for (int r = 0; r < 8; ++r) gw[r] = gs[(wm * 16 + r + 8 * hlf) * E_NUM + e];
#pragma unroll
    for (int t = 0; t < 8; ++t) {
      const float bias2 = b2[e * O_DIM + wn * 128 + t * 16 + nloc];
#pragma unroll
      for (int r = 0; r < 8; ++r)
        outacc[t][r] += gw[r] * (eoacc[t][r] + bias2);
    }
  }

  // ---- writeback [BN x 1024] f32
#pragma unroll
  for (int t = 0; t < 8; ++t) {
    const int col = wn * 128 + t * 16 + nloc;
#pragma unroll
    for (int r = 0; r < 8; ++r) {
      const int m = wm * 16 + r + 8 * hlf;
      out[(long)(tile0 + m) * O_DIM + col] = outacc[t][r];
    }
  }
}

// ---------------------------------------------------------------------------
extern "C" void kernel_launch(void* const* d_in, const int* in_sizes, int n_in,
                              void* d_out, int out_size, void* d_ws, size_t ws_size,
                              hipStream_t stream) {
  const float* x  = (const float*)d_in[0];
  const float* w1 = (const float*)d_in[1];
  const float* b1 = (const float*)d_in[2];
  const float* w2 = (const float*)d_in[3];
  const float* b2 = (const float*)d_in[4];
  const float* wg = (const float*)d_in[5];
  const float* bg = (const float*)d_in[6];
  float* out = (float*)d_out;

  char* ws = (char*)d_ws;
  __bf16* xb   = (__bf16*)ws;
  __bf16* w1p  = (__bf16*)(ws + (size_t)N_TOK * D_IN * 2);
  __bf16* w2p  = w1p + (size_t)E_NUM * D_IN * H_DIM;
  float*  gate = (float*)((char*)(w2p + (size_t)E_NUM * H_DIM * O_DIM));

  long nx = (long)N_TOK * D_IN;
  convert_x_bf16<<<(int)((nx + 255) / 256), 256, 0, stream>>>(x, xb, nx);

  long nw1 = (long)E_NUM * D_IN * H_DIM;
  pack_weights_bf16<<<(int)((nw1 + 255) / 256), 256, 0, stream>>>(w1, w1p, D_IN, H_DIM);

  long nw2 = (long)E_NUM * H_DIM * O_DIM;
  pack_weights_bf16<<<(int)((nw2 + 255) / 256), 256, 0, stream>>>(w2, w2p, H_DIM, O_DIM);

  gating_kernel<<<N_TOK / 8, 256, 0, stream>>>(x, wg, bg, gate);

  moe_ffn_kernel<<<N_TOK / BN, 512, LDS_TOTAL, stream>>>(xb, w1p, w2p, b1, b2, gate, out);
}
